// multicsLoss_50225347560090
// MI455X (gfx1250) — matrix operand order repaired and checked by reading, not compiled, not run on hardware
//
#include <hip/hip_runtime.h>
#include <math.h>

// ---------------------------------------------------------------------------
// Fused RBF-MMD style loss for MI455X (gfx1250, wave32, WMMA).
//
// Six Gram means over 8192x8192 tiles, fused GEMM(bf16 split-3) + exp + sum.
// A/B fragments loaded from LDS per CDNA5 16-bit WMMA VGPR layouts:
//   A (16x32): lanes 0-15 hold M=lane, K = {k0..k0+7} u {k0+16..k0+23}, k0 = 0
//              lanes 16-31 same rows, k0 = 8   -> two 16B contiguous runs
//   B (32x16): lanes 0-15 hold N=lane, same K split as A (k0 = 0 / 8)
// C/D (16x16 f32): VGPR v -> M = v + 8*(lane>=16), N = lane%16.
// ---------------------------------------------------------------------------

typedef __bf16 bf16;
typedef __attribute__((ext_vector_type(8)))  bf16  v8bf;
typedef __attribute__((ext_vector_type(16))) bf16  v16bf;
typedef __attribute__((ext_vector_type(8)))  float v8f;

#define NPTS  8192
#define DDIM  64
#define BT    128          // output tile edge
#define LDSS  72           // padded LDS row stride (bf16 elems) -> conflict-free
#define NTILE (NPTS / BT)  // 64
#define NBLK  (NTILE * NTILE) // 4096 partials per pair

__device__ __forceinline__ v16bf combine8(v8bf lo, v8bf hi) {
  v16bf r;
#pragma unroll
  for (int i = 0; i < 8; ++i) { r[i] = lo[i]; r[i + 8] = hi[i]; }
  return r;
}

__global__ __launch_bounds__(256)
void rbf_tile_kernel(const float* __restrict__ X,
                     const float* __restrict__ Y,
                     const float* __restrict__ Z,
                     float* __restrict__ partials)
{
  __shared__ bf16 Ahi[BT][LDSS];
  __shared__ bf16 Alo[BT][LDSS];
  __shared__ bf16 Bhi[BT][LDSS];
  __shared__ bf16 Blo[BT][LDSS];
  __shared__ float nrmA[BT];
  __shared__ float nrmB[BT];
  __shared__ float red[256];

  // pair selection: 0:XX 1:ZZ 2:XZ 3:XY 4:YY 5:YZ
  const int p = blockIdx.z;
  const float* Aptr;
  const float* Bptr;
  switch (p) {
    case 0:  Aptr = X; Bptr = X; break;
    case 1:  Aptr = Z; Bptr = Z; break;
    case 2:  Aptr = X; Bptr = Z; break;
    case 3:  Aptr = X; Bptr = Y; break;
    case 4:  Aptr = Y; Bptr = Y; break;
    default: Aptr = Y; Bptr = Z; break;
  }
  const int rowBase = blockIdx.y * BT;
  const int colBase = blockIdx.x * BT;
  const int tid = threadIdx.x;

  // ---- cooperative global->LDS load with f32 -> (bf16 hi, bf16 lo) split ----
#pragma unroll
  for (int i = 0; i < (BT * DDIM) / 256; ++i) {   // 32 iters, coalesced
    int e = tid + i * 256;
    int r = e >> 6;        // /64
    int c = e & 63;
    float a  = Aptr[(size_t)(rowBase + r) * DDIM + c];
    bf16 ah  = (bf16)a;
    bf16 al  = (bf16)(a - (float)ah);
    Ahi[r][c] = ah;  Alo[r][c] = al;
    float b  = Bptr[(size_t)(colBase + r) * DDIM + c];
    bf16 bh  = (bf16)b;
    bf16 bl  = (bf16)(b - (float)bh);
    Bhi[r][c] = bh;  Blo[r][c] = bl;
  }
  // exact fp32 row norms (data is L2-resident; 64 fmas per thread)
  if (tid < BT) {
    const float* rp = Aptr + (size_t)(rowBase + tid) * DDIM;
    float s = 0.f;
#pragma unroll
    for (int k = 0; k < DDIM; ++k) { float v = rp[k]; s = fmaf(v, v, s); }
    nrmA[tid] = s;
  } else {
    const int t = tid - BT;
    const float* rp = Bptr + (size_t)(colBase + t) * DDIM;
    float s = 0.f;
#pragma unroll
    for (int k = 0; k < DDIM; ++k) { float v = rp[k]; s = fmaf(v, v, s); }
    nrmB[t] = s;
  }
  __syncthreads();

  // ---- wave tiling: 8 waves as 4x2; each wave owns 32(M) x 64(N) ----
  const int lane  = tid & 31;
  const int w     = tid >> 5;
  const int wr    = w >> 1;       // 0..3
  const int wc    = w & 1;        // 0..1
  const int lhalf = lane >> 4;    // 0/1
  const int l16   = lane & 15;

  v8f acc[2][4];
#pragma unroll
  for (int i = 0; i < 2; ++i)
#pragma unroll
    for (int j = 0; j < 4; ++j)
#pragma unroll
      for (int v = 0; v < 8; ++v) acc[i][j][v] = 0.f;

#pragma unroll
  for (int kk = 0; kk < 2; ++kk) {       // two K=32 chunks cover D=64
    const int kb = kk * 32;

    v16bf aH[2], aL[2];
#pragma unroll
    for (int i = 0; i < 2; ++i) {
      const int row = wr * 32 + i * 16 + l16;
      const int k0  = kb + lhalf * 8;
      aH[i] = combine8(*(const v8bf*)&Ahi[row][k0],
                       *(const v8bf*)&Ahi[row][k0 + 16]);
      aL[i] = combine8(*(const v8bf*)&Alo[row][k0],
                       *(const v8bf*)&Alo[row][k0 + 16]);
    }
    v16bf bH[4], bL[4];
#pragma unroll
    for (int j = 0; j < 4; ++j) {
      const int col = wc * 64 + j * 16 + l16;
      const int k0  = kb + lhalf * 8;
      bH[j] = combine8(*(const v8bf*)&Bhi[col][k0],
                       *(const v8bf*)&Bhi[col][k0 + 16]);
      bL[j] = combine8(*(const v8bf*)&Blo[col][k0],
                       *(const v8bf*)&Blo[col][k0 + 16]);
    }
#pragma unroll
    for (int i = 0; i < 2; ++i) {
#pragma unroll
      for (int j = 0; j < 4; ++j) {
        // 3xBF16 split: hi*hi + hi*lo + lo*hi, f32 accumulate
        acc[i][j] = __builtin_amdgcn_wmma_f32_16x16x32_bf16(
            false, aH[i], false, bH[j], (short)0, acc[i][j], false, false);
        acc[i][j] = __builtin_amdgcn_wmma_f32_16x16x32_bf16(
            false, aH[i], false, bL[j], (short)0, acc[i][j], false, false);
        acc[i][j] = __builtin_amdgcn_wmma_f32_16x16x32_bf16(
            false, aL[i], false, bH[j], (short)0, acc[i][j], false, false);
      }
    }
  }

  // ---- epilogue: sq = |a|^2 + |b|^2 - 2ab ; sum exp(-sq / KSIZE) ----
  const float invks = 0.5f;   // 1 / KSIZE, KSIZE = 2
  float tsum = 0.f;
#pragma unroll
  for (int i = 0; i < 2; ++i) {
#pragma unroll
    for (int j = 0; j < 4; ++j) {
      const int colN = wc * 64 + j * 16 + l16;
      const float nb = nrmB[colN];
#pragma unroll
      for (int v = 0; v < 8; ++v) {
        const int rowM = wr * 32 + i * 16 + lhalf * 8 + v;
        const float sq = nrmA[rowM] + nb - 2.0f * acc[i][j][v];
        tsum += __expf(-sq * invks);
      }
    }
  }

  // ---- deterministic fixed-order block reduction ----
  red[tid] = tsum;
  __syncthreads();
#pragma unroll
  for (int s = 128; s > 0; s >>= 1) {
    if (tid < s) red[tid] += red[tid + s];
    __syncthreads();
  }
  if (tid == 0)
    partials[(size_t)p * NBLK + blockIdx.y * NTILE + blockIdx.x] = red[0];
}

__global__ __launch_bounds__(256)
void rbf_finalize_kernel(const float* __restrict__ partials,
                         float* __restrict__ out)
{
  __shared__ float red[256];
  __shared__ float means[6];
  const int tid = threadIdx.x;

  for (int p = 0; p < 6; ++p) {
    float s = 0.f;
    for (int i = tid; i < NBLK; i += 256)        // fixed-order strided sum
      s += partials[(size_t)p * NBLK + i];
    red[tid] = s;
    __syncthreads();
    for (int st = 128; st > 0; st >>= 1) {
      if (tid < st) red[tid] += red[tid + st];
      __syncthreads();
    }
    if (tid == 0) {
      const double normc = sqrt(2.0 * 3.14159265358979323846 * 2.0); // sqrt(2*pi*ksize)
      means[p] = (float)((double)red[0] /
                         ((double)NPTS * (double)NPTS) / normc);
    }
    __syncthreads();
  }

  if (tid == 0) {
    const float SIGMA = 3.0f, EPS = 1e-5f;
    const float mxx = means[0], mzz = means[1], mxz = means[2];
    const float mxy = means[3], myy = means[4], myz = means[5];
    const float r1 = logf(SIGMA * sqrtf(mxx * mzz + EPS) / (mxz + EPS));
    const float r2 = logf(SIGMA * sqrtf(myy * mzz + EPS) / (myz + EPS));
    const float r3 = logf(SIGMA * sqrtf(mxx * myy + EPS) / (mxy + EPS));
    out[0] = r1 * 10.0f + r2 + r3 * 10.0f;
  }
}

extern "C" void kernel_launch(void* const* d_in, const int* in_sizes, int n_in,
                              void* d_out, int out_size, void* d_ws, size_t ws_size,
                              hipStream_t stream) {
  (void)in_sizes; (void)n_in; (void)out_size; (void)ws_size;
  const float* X = (const float*)d_in[0];
  const float* Y = (const float*)d_in[1];
  const float* Z = (const float*)d_in[2];
  float* partials = (float*)d_ws;          // 6 * 4096 floats = 96 KB

  dim3 grid(NTILE, NTILE, 6);
  rbf_tile_kernel<<<grid, dim3(256), 0, stream>>>(X, Y, Z, partials);
  rbf_finalize_kernel<<<1, dim3(256), 0, stream>>>(partials, (float*)d_out);
}
